// Model_68186900792241
// MI455X (gfx1250) — matrix operand order repaired and checked by reading, not compiled, not run on hardware
//
#include <hip/hip_runtime.h>

typedef __attribute__((ext_vector_type(16))) _Float16 v16h;
typedef __attribute__((ext_vector_type(8)))  _Float16 h8;
typedef __attribute__((ext_vector_type(4)))  _Float16 h4;
typedef __attribute__((ext_vector_type(8)))  float    v8f;
typedef __attribute__((ext_vector_type(4)))  float    f4;

#define SEQ 2048
#define HD  128
#define KT  32            // keys per LDS tile
#define NT  (SEQ / KT)    // 64 tiles

static __device__ __forceinline__ v16h cat16(h8 a, h8 b) {
    return __builtin_shufflevector(a, b, 0,1,2,3,4,5,6,7,8,9,10,11,12,13,14,15);
}

__launch_bounds__(256)
__global__ void fa_wmma_kernel(const float* __restrict__ Q,
                               const float* __restrict__ K,
                               const float* __restrict__ V,
                               float* __restrict__ O)
{
    // K tile: 32 keys x 128 d, f16, row-major (+8 pad halfs vs bank conflicts)
    __shared__ __align__(16) _Float16 sK[KT][HD + 8];
    // V tile transposed: 128 d x 32 keys, f16 (+8 pad)
    __shared__ __align__(16) _Float16 sVT[HD][KT + 8];

    const int tid  = threadIdx.x;
    const int lane = tid & 31;
    const int wave = tid >> 5;
    const int n    = lane & 15;   // query column owned by this lane (C layout N)
    const int hi   = lane >> 4;   // lane half

    const int bh = blockIdx.x >> 4;          // 64 (batch*heads)
    const int qt = blockIdx.x & 15;          // 16 query tiles of 128 per head
    const int q0 = qt * 128 + wave * 16;     // this wave's 16 query rows
    const size_t headOff = (size_t)bh * SEQ * HD;

    const float SCALE = 0.08838834764831845f * 1.44269504f; // rsqrt(128)*log2e

    // ---- preload Q^T as 4 B-fragments (32 d each), scaled ----
    // B layout (32x16 f16): lane holds N = lane%16, element e -> K(d) = c*32 + hi*16 + e
    v16h qf[4];
    {
        const float* qrow = Q + headOff + (size_t)(q0 + n) * HD;
        #pragma unroll
        for (int c = 0; c < 4; ++c) {
            const f4* p = (const f4*)(qrow + c * 32 + hi * 16);
            f4 x0 = p[0], x1 = p[1], x2 = p[2], x3 = p[3];
            v16h qh;
            #pragma unroll
            for (int i = 0; i < 4; ++i) {
                qh[i]      = (_Float16)(x0[i] * SCALE);
                qh[4 + i]  = (_Float16)(x1[i] * SCALE);
                qh[8 + i]  = (_Float16)(x2[i] * SCALE);
                qh[12 + i] = (_Float16)(x3[i] * SCALE);
            }
            qf[c] = qh;
        }
    }

    v8f o[8] = {};                 // O^T accumulators: 8 d-tiles of 16x16 f32
    float m = -3.0e38f, l = 0.0f;  // online-softmax stats for query n (both halves)

    // staging coordinates (256 threads)
    const int krow = tid >> 3, kc0 = (tid & 7) * 16;       // K: 16 floats per thread
    const int vr0 = (tid >> 5) * 4, vc0 = (tid & 31) * 4;  // V: 4x4 block per thread

    for (int it = 0; it < NT; ++it) {
        const int kb = it * KT;

        // ---- stage K tile -> LDS f16 ----
        {
            const f4* src = (const f4*)(K + headOff + (size_t)(kb + krow) * HD + kc0);
            f4 a = src[0], b = src[1], c = src[2], d = src[3];
            h8 h0, h1;
            #pragma unroll
            for (int i = 0; i < 4; ++i) {
                h0[i] = (_Float16)a[i]; h0[4 + i] = (_Float16)b[i];
                h1[i] = (_Float16)c[i]; h1[4 + i] = (_Float16)d[i];
            }
            *(h8*)&sK[krow][kc0]     = h0;
            *(h8*)&sK[krow][kc0 + 8] = h1;
        }
        // ---- stage V tile transposed -> LDS f16 ----
        {
            const float* vb = V + headOff + (size_t)(kb + vr0) * HD + vc0;
            f4 a0 = *(const f4*)(vb);
            f4 a1 = *(const f4*)(vb + HD);
            f4 a2 = *(const f4*)(vb + 2 * HD);
            f4 a3 = *(const f4*)(vb + 3 * HD);
            #pragma unroll
            for (int i = 0; i < 4; ++i) {
                h4 t;
                t[0] = (_Float16)a0[i]; t[1] = (_Float16)a1[i];
                t[2] = (_Float16)a2[i]; t[3] = (_Float16)a3[i];
                *(h4*)&sVT[vc0 + i][vr0] = t;
            }
        }
        __syncthreads();

        // prefetch next tile into caches (global_prefetch_b8)
        if (it + 1 < NT) {
            __builtin_prefetch(K + headOff + (size_t)(kb + KT + krow) * HD + kc0, 0, 0);
            __builtin_prefetch(V + headOff + (size_t)(kb + KT + vr0) * HD + vc0, 0, 0);
        }

        // ---- S^T = K_tile x Q^T : two 16-key sub-tiles, 4 d-chunks each ----
        // A layout (16x32 f16): lane row M = n, element e -> K = hi*16 + e
        v8f st0 = {}, st1 = {};
        #pragma unroll
        for (int c = 0; c < 4; ++c) {
            h8 a_lo = *(const h8*)&sK[n][c * 32 + hi * 16];
            h8 a_hi = *(const h8*)&sK[n][c * 32 + hi * 16 + 8];
            v16h a = cat16(a_lo, a_hi);
            st0 = __builtin_amdgcn_wmma_f32_16x16x32_f16(false, a, false, qf[c],
                                                         (short)0, st0, false, false);
        }
        #pragma unroll
        for (int c = 0; c < 4; ++c) {
            h8 a_lo = *(const h8*)&sK[16 + n][c * 32 + hi * 16];
            h8 a_hi = *(const h8*)&sK[16 + n][c * 32 + hi * 16 + 8];
            v16h a = cat16(a_lo, a_hi);
            st1 = __builtin_amdgcn_wmma_f32_16x16x32_f16(false, a, false, qf[c],
                                                         (short)0, st1, false, false);
        }

        // ---- online softmax over this tile's 32 keys (query = n, lane-local) ----
        float mt = st0[0];
        #pragma unroll
        for (int r = 1; r < 8; ++r) mt = fmaxf(mt, st0[r]);
        #pragma unroll
        for (int r = 0; r < 8; ++r) mt = fmaxf(mt, st1[r]);
        mt = fmaxf(mt, __shfl_xor(mt, 16, 32));
        const float mnew  = fmaxf(m, mt);
        const float alpha = __expf(m - mnew);
        float p0[8], p1[8], rs = 0.0f;
        #pragma unroll
        for (int r = 0; r < 8; ++r) {
            p0[r] = __expf(st0[r] - mnew);
            p1[r] = __expf(st1[r] - mnew);
            rs += p0[r] + p1[r];
        }
        rs += __shfl_xor(rs, 16, 32);
        l = l * alpha + rs;
        m = mnew;
        #pragma unroll
        for (int dt = 0; dt < 8; ++dt) o[dt] *= alpha;

        // ---- build P^T B-fragment (32 keys x 16 queries) ----
        // lo lane: e<8 own st0 rows 0..7, e>=8 partner st0 rows 8..15
        // hi lane: e<8 partner st1 rows 0..7 (keys16..23), e>=8 own st1 (keys24..31)
        v16h pb;
        #pragma unroll
        for (int r = 0; r < 8; ++r) {
            float x0 = __shfl_xor(p0[r], 16, 32);
            float x1 = __shfl_xor(p1[r], 16, 32);
            pb[r]     = (_Float16)(hi ? x1 : p0[r]);
            pb[8 + r] = (_Float16)(hi ? p1[r] : x0);
        }

        // ---- O^T += V^T x P^T : 8 d-tiles ----
        #pragma unroll
        for (int dt = 0; dt < 8; ++dt) {
            h8 a_lo = *(const h8*)&sVT[dt * 16 + n][hi * 16];
            h8 a_hi = *(const h8*)&sVT[dt * 16 + n][hi * 16 + 8];
            v16h a = cat16(a_lo, a_hi);
            o[dt] = __builtin_amdgcn_wmma_f32_16x16x32_f16(false, a, false, pb,
                                                           (short)0, o[dt], false, false);
        }
        __syncthreads();
    }

    // ---- epilogue: normalize and store O (fp32) ----
    const float inv = 1.0f / l;
    float* orow = O + headOff + (size_t)(q0 + n) * HD;
    #pragma unroll
    for (int dt = 0; dt < 8; ++dt) {
        f4 y0 = { o[dt][0] * inv, o[dt][1] * inv, o[dt][2] * inv, o[dt][3] * inv };
        f4 y1 = { o[dt][4] * inv, o[dt][5] * inv, o[dt][6] * inv, o[dt][7] * inv };
        *(f4*)(orow + dt * 16 + 8 * hi)     = y0;
        *(f4*)(orow + dt * 16 + 8 * hi + 4) = y1;
    }
}

extern "C" void kernel_launch(void* const* d_in, const int* in_sizes, int n_in,
                              void* d_out, int out_size, void* d_ws, size_t ws_size,
                              hipStream_t stream) {
    (void)in_sizes; (void)n_in; (void)out_size; (void)d_ws; (void)ws_size;
    const float* Q = (const float*)d_in[0];
    const float* K = (const float*)d_in[1];
    const float* V = (const float*)d_in[2];
    float* O = (float*)d_out;
    // grid: 64 (batch*head) x 16 query-tiles of 128 rows; 256 threads = 8 wave32
    fa_wmma_kernel<<<dim3(64 * 16), dim3(256), 0, stream>>>(Q, K, V, O);
}
